// AttentionHead_30863634989138
// MI455X (gfx1250) — matrix operand order repaired and checked by reading, not compiled, not run on hardware
//
#include <hip/hip_runtime.h>

// ---------------------------------------------------------------------------
// AttentionHead on MI455X (gfx1250): bf16 WMMA everywhere.
//   B=8, T=2048, E=H=1024.  Pipeline:
//     cvt(f32->bf16) -> Q/K GEMM, V GEMM (transposed out)
//     -> causal online-softmax stats + P (bf16) -> O = P*V (f32 out)
// Wave tiles are 32x64 (2x4 WMMA tiles); per K-step all fragment loads are
// hoisted ahead of the 8-WMMA burst.  No explicit double buffering: that
// spills (fragments + 64 acc VGPRs exceed budget); cross-wave occupancy
// (8 waves/block) hides the per-fragment load latency instead.
// ---------------------------------------------------------------------------

typedef __bf16 bf16;
typedef __attribute__((ext_vector_type(16))) __bf16 v16bf;
typedef __attribute__((ext_vector_type(8)))  __bf16 v8bf;
typedef __attribute__((ext_vector_type(8)))  float  v8f;

#define BDIM 8
#define TDIM 2048
#define EDIM 1024
#define HDIM 1024
#define BT   (BDIM * TDIM)

// ---- WMMA fragment helpers (ISA 7.12.2 layouts, wave32) --------------------

// A-matrix 16x32 bf16, row-major source with leading dim `ld`.
// lanes 0-15: row M=lane, K chunks {k0..k0+7, k0+16..k0+23}
// lanes 16-31: row M=lane-16, K chunks {k0+8..k0+15, k0+24..k0+31}
__device__ __forceinline__ v16bf load_a_frag(const bf16* base, int ld,
                                             int m0, int k0, int lane) {
  const int half = lane >> 4, l = lane & 15;
  const bf16* p = base + (size_t)(m0 + l) * ld + k0 + half * 8;
  v8bf lo = *(const v8bf*)(p);
  v8bf hi = *(const v8bf*)(p + 16);
  v16bf r;
#pragma unroll
  for (int i = 0; i < 8; ++i) { r[i] = lo[i]; r[8 + i] = hi[i]; }
  return r;
}

// B-matrix 32x16 bf16 taken from an [N, K] row-major source (lane = column N,
// lanes 0-15 -> K chunks {k0..k0+7, k0+16..k0+23}, lanes 16-31 -> +8).
__device__ __forceinline__ v16bf load_b_frag(const bf16* base, int ld,
                                             int n0, int k0, int lane) {
  const int half = lane >> 4, l = lane & 15;
  const bf16* p = base + (size_t)(n0 + l) * ld + k0 + half * 8;
  v8bf lo = *(const v8bf*)(p);
  v8bf hi = *(const v8bf*)(p + 16);
  v16bf r;
#pragma unroll
  for (int i = 0; i < 8; ++i) { r[i] = lo[i]; r[8 + i] = hi[i]; }
  return r;
}

__device__ __forceinline__ v8f wmma_bf16(v16bf a, v16bf b, v8f c) {
  return __builtin_amdgcn_wmma_f32_16x16x32_bf16(
      false, a, false, b, (short)0, c, false, false);
}

// ---- kernel 0: f32 -> bf16 --------------------------------------------------

__global__ void cvt_f32_bf16(const float* __restrict__ src,
                             bf16* __restrict__ dst, size_t n) {
  size_t i = (size_t)blockIdx.x * blockDim.x + threadIdx.x;
  size_t stride = (size_t)gridDim.x * blockDim.x;
  for (; i < n; i += stride) dst[i] = (bf16)src[i];
}

// ---- kernel 1: projection GEMM  C[M,N] = scale * X[M,K] * W[N,K]^T ----------
// 256 thr = 8 waves (4M x 2N); wave tile 32x64; block tile 128x128.
// trans==1 stores C transposed per batch as [B][N][T] (for V).

__global__ void proj_gemm(const bf16* __restrict__ X, const bf16* __restrict__ W,
                          bf16* __restrict__ C, float scale, int trans) {
  const int lane = threadIdx.x & 31;
  const int wid  = threadIdx.x >> 5;
  const int m0 = blockIdx.x * 128 + (wid >> 1) * 32;
  const int n0 = blockIdx.y * 128 + (wid & 1) * 64;

  v8f z = {};
  v8f acc[2][4];
#pragma unroll
  for (int i = 0; i < 2; ++i)
#pragma unroll
    for (int j = 0; j < 4; ++j) acc[i][j] = z;

  for (int k = 0; k < EDIM; k += 32) {
    __builtin_prefetch(X + (size_t)(m0 + (lane & 15)) * EDIM + k + 32, 0, 3);
    v16bf a0 = load_a_frag(X, EDIM, m0,      k, lane);
    v16bf a1 = load_a_frag(X, EDIM, m0 + 16, k, lane);
    v16bf b0 = load_b_frag(W, EDIM, n0,      k, lane);
    v16bf b1 = load_b_frag(W, EDIM, n0 + 16, k, lane);
    v16bf b2 = load_b_frag(W, EDIM, n0 + 32, k, lane);
    v16bf b3 = load_b_frag(W, EDIM, n0 + 48, k, lane);
    acc[0][0] = wmma_bf16(a0, b0, acc[0][0]);
    acc[1][0] = wmma_bf16(a1, b0, acc[1][0]);
    acc[0][1] = wmma_bf16(a0, b1, acc[0][1]);
    acc[1][1] = wmma_bf16(a1, b1, acc[1][1]);
    acc[0][2] = wmma_bf16(a0, b2, acc[0][2]);
    acc[1][2] = wmma_bf16(a1, b2, acc[1][2]);
    acc[0][3] = wmma_bf16(a0, b3, acc[0][3]);
    acc[1][3] = wmma_bf16(a1, b3, acc[1][3]);
  }

  const int half = lane >> 4, l = lane & 15;
  if (!trans) {
#pragma unroll
    for (int i = 0; i < 2; ++i)
#pragma unroll
      for (int j = 0; j < 4; ++j) {
        int row = m0 + i * 16 + half * 8;
        int col = n0 + j * 16 + l;
#pragma unroll
        for (int r = 0; r < 8; ++r)
          C[(size_t)(row + r) * HDIM + col] = (bf16)(acc[i][j][r] * scale);
      }
  } else {
#pragma unroll
    for (int i = 0; i < 2; ++i)
#pragma unroll
      for (int j = 0; j < 4; ++j) {
        int rowg = m0 + i * 16 + half * 8;   // global t index (b*T + t)
        int col  = n0 + j * 16 + l;          // h index
        int bb = rowg / TDIM, t = rowg % TDIM;
        v8bf pk;
#pragma unroll
        for (int r = 0; r < 8; ++r) pk[r] = (bf16)(acc[i][j][r] * scale);
        *(v8bf*)(C + ((size_t)bb * HDIM + col) * TDIM + t) = pk;
      }
  }
}

// ---- kernel 2: causal scores + online softmax -> P (bf16) -------------------
// Block 128 thr = 4 waves; each wave owns 16 query rows and sweeps 64 keys per
// iteration (1 A-frag amortized over 4 B-frags).  E^-0.5 prefolded into Q.

__device__ __forceinline__ void qk_tile4(const bf16* Qb, const bf16* Kb,
                                         int m0, int s0, int lane, v8f acc[4]) {
  v8f z = {};
#pragma unroll
  for (int j = 0; j < 4; ++j) acc[j] = z;
  for (int k = 0; k < HDIM; k += 32) {
    __builtin_prefetch(Qb + (size_t)(m0 + (lane & 15)) * HDIM + k + 32, 0, 3);
    v16bf a  = load_a_frag(Qb, HDIM, m0, k, lane);
    v16bf b0 = load_b_frag(Kb, HDIM, s0,      k, lane);
    v16bf b1 = load_b_frag(Kb, HDIM, s0 + 16, k, lane);
    v16bf b2 = load_b_frag(Kb, HDIM, s0 + 32, k, lane);
    v16bf b3 = load_b_frag(Kb, HDIM, s0 + 48, k, lane);
    acc[0] = wmma_bf16(a, b0, acc[0]);
    acc[1] = wmma_bf16(a, b1, acc[1]);
    acc[2] = wmma_bf16(a, b2, acc[2]);
    acc[3] = wmma_bf16(a, b3, acc[3]);
  }
}

__global__ void attn_softmax_p(const bf16* __restrict__ Q,
                               const bf16* __restrict__ K,
                               bf16* __restrict__ P) {
  const int b    = blockIdx.y;
  const int lane = threadIdx.x & 31;
  const int wid  = threadIdx.x >> 5;
  const int m0   = blockIdx.x * 64 + wid * 16;

  const bf16* Qb = Q + (size_t)b * TDIM * HDIM;
  const bf16* Kb = K + (size_t)b * TDIM * HDIM;
  bf16*       Pb = P + (size_t)b * TDIM * TDIM;

  const int half = lane >> 4, l = lane & 15;
  const int rowb = m0 + half * 8;          // first row this lane-half holds

  float mrun[8], lrun[8];
#pragma unroll
  for (int r = 0; r < 8; ++r) { mrun[r] = -3.0e38f; lrun[r] = 0.0f; }

  // ---- sweep 1: running (max, sumexp) per row -----------------------------
  for (int s0 = 0; s0 <= m0 + 15; s0 += 64) {
    v8f acc[4];
    qk_tile4(Qb, Kb, m0, s0, lane, acc);
#pragma unroll
    for (int r = 0; r < 8; ++r) {
      const int row = rowb + r;
      float v[4], mx = -3.0e38f;
#pragma unroll
      for (int j = 0; j < 4; ++j) {
        int c = s0 + j * 16 + l;
        v[j] = (c <= row) ? acc[j][r] : -3.0e38f;
        mx = fmaxf(mx, v[j]);
      }
#pragma unroll
      for (int off = 1; off < 16; off <<= 1)
        mx = fmaxf(mx, __shfl_xor(mx, off, 32));   // stays within 16-lane half
      float mnew = fmaxf(mrun[r], mx);
      float sm = 0.0f;
#pragma unroll
      for (int j = 0; j < 4; ++j) {
        int c = s0 + j * 16 + l;
        sm += (c <= row) ? __expf(v[j] - mnew) : 0.0f;
      }
#pragma unroll
      for (int off = 1; off < 16; off <<= 1)
        sm += __shfl_xor(sm, off, 32);
      lrun[r] = lrun[r] * __expf(mrun[r] - mnew) + sm;
      mrun[r] = mnew;
    }
  }

  // ---- sweep 2: recompute tiles, emit normalized P ------------------------
  for (int s0 = 0; s0 <= m0 + 15; s0 += 64) {
    v8f acc[4];
    qk_tile4(Qb, Kb, m0, s0, lane, acc);
#pragma unroll
    for (int r = 0; r < 8; ++r) {
      const int row = rowb + r;
      float inv = 1.0f / lrun[r];
#pragma unroll
      for (int j = 0; j < 4; ++j) {
        int c = s0 + j * 16 + l;
        float p = (c <= row) ? __expf(acc[j][r] - mrun[r]) * inv : 0.0f;
        Pb[(size_t)row * TDIM + c] = (bf16)p;
      }
    }
  }
}

// ---- kernel 3: O[b,t,h] = sum_s P[b,t,s] * V[b,s,h]  (V stored as [B][H][T])
// 256 thr = 8 waves (4M x 2N); wave tile 32x64; causal K-loop bound.

__global__ void attn_pv(const bf16* __restrict__ P, const bf16* __restrict__ Vt,
                        float* __restrict__ O) {
  const int b    = blockIdx.z;
  const int lane = threadIdx.x & 31;
  const int wid  = threadIdx.x >> 5;
  const int m0 = blockIdx.x * 128 + (wid >> 1) * 32;
  const int n0 = blockIdx.y * 128 + (wid & 1) * 64;

  const bf16* Pb = P  + (size_t)b * TDIM * TDIM;
  const bf16* Vb = Vt + (size_t)b * HDIM * TDIM;
  float*      Ob = O  + (size_t)b * TDIM * HDIM;

  v8f z = {};
  v8f acc[2][4];
#pragma unroll
  for (int i = 0; i < 2; ++i)
#pragma unroll
    for (int j = 0; j < 4; ++j) acc[i][j] = z;

  for (int s0 = 0; s0 <= m0 + 31; s0 += 32) {   // causal: P is zero beyond
    __builtin_prefetch(Pb + (size_t)(m0 + (lane & 15)) * TDIM + s0 + 32, 0, 3);
    v16bf a0 = load_a_frag(Pb, TDIM, m0,      s0, lane);
    v16bf a1 = load_a_frag(Pb, TDIM, m0 + 16, s0, lane);
    v16bf b0 = load_b_frag(Vb, TDIM, n0,      s0, lane);
    v16bf b1 = load_b_frag(Vb, TDIM, n0 + 16, s0, lane);
    v16bf b2 = load_b_frag(Vb, TDIM, n0 + 32, s0, lane);
    v16bf b3 = load_b_frag(Vb, TDIM, n0 + 48, s0, lane);
    acc[0][0] = wmma_bf16(a0, b0, acc[0][0]);
    acc[1][0] = wmma_bf16(a1, b0, acc[1][0]);
    acc[0][1] = wmma_bf16(a0, b1, acc[0][1]);
    acc[1][1] = wmma_bf16(a1, b1, acc[1][1]);
    acc[0][2] = wmma_bf16(a0, b2, acc[0][2]);
    acc[1][2] = wmma_bf16(a1, b2, acc[1][2]);
    acc[0][3] = wmma_bf16(a0, b3, acc[0][3]);
    acc[1][3] = wmma_bf16(a1, b3, acc[1][3]);
  }

  const int half = lane >> 4, l = lane & 15;
#pragma unroll
  for (int i = 0; i < 2; ++i)
#pragma unroll
    for (int j = 0; j < 4; ++j) {
      int row = m0 + i * 16 + half * 8;
      int col = n0 + j * 16 + l;
#pragma unroll
      for (int r = 0; r < 8; ++r)
        Ob[(size_t)(row + r) * HDIM + col] = acc[i][j][r];
    }
}

// ---------------------------------------------------------------------------

extern "C" void kernel_launch(void* const* d_in, const int* in_sizes, int n_in,
                              void* d_out, int out_size, void* d_ws, size_t ws_size,
                              hipStream_t stream) {
  const float* x  = (const float*)d_in[0];
  const float* Wk = (const float*)d_in[1];
  const float* Wq = (const float*)d_in[2];
  const float* Wv = (const float*)d_in[3];
  float* out = (float*)d_out;

  char* ws = (char*)d_ws;
  bf16* xb  = (bf16*)ws; ws += (size_t)BT * EDIM * sizeof(bf16);
  bf16* Wkb = (bf16*)ws; ws += (size_t)HDIM * EDIM * sizeof(bf16);
  bf16* Wqb = (bf16*)ws; ws += (size_t)HDIM * EDIM * sizeof(bf16);
  bf16* Wvb = (bf16*)ws; ws += (size_t)HDIM * EDIM * sizeof(bf16);
  bf16* Qb  = (bf16*)ws; ws += (size_t)BT * HDIM * sizeof(bf16);
  bf16* Kb  = (bf16*)ws; ws += (size_t)BT * HDIM * sizeof(bf16);
  bf16* Vtb = (bf16*)ws; ws += (size_t)BDIM * HDIM * TDIM * sizeof(bf16);
  bf16* Pb  = (bf16*)ws; ws += (size_t)BDIM * TDIM * TDIM * sizeof(bf16);
  (void)ws_size; (void)in_sizes; (void)n_in; (void)out_size;

  cvt_f32_bf16<<<4096, 256, 0, stream>>>(x,  xb,  (size_t)BT * EDIM);
  cvt_f32_bf16<<<512,  256, 0, stream>>>(Wk, Wkb, (size_t)HDIM * EDIM);
  cvt_f32_bf16<<<512,  256, 0, stream>>>(Wq, Wqb, (size_t)HDIM * EDIM);
  cvt_f32_bf16<<<512,  256, 0, stream>>>(Wv, Wvb, (size_t)HDIM * EDIM);

  const float qscale = 0.03125f;  // 1024^-0.5
  dim3 gproj(BT / 128, HDIM / 128);
  proj_gemm<<<gproj, 256, 0, stream>>>(xb, Wqb, Qb,  qscale, 0);
  proj_gemm<<<gproj, 256, 0, stream>>>(xb, Wkb, Kb,  1.0f,   0);
  proj_gemm<<<gproj, 256, 0, stream>>>(xb, Wvb, Vtb, 1.0f,   1);

  attn_softmax_p<<<dim3(TDIM / 64, BDIM), 128, 0, stream>>>(Qb, Kb, Pb);

  attn_pv<<<dim3(TDIM / 128, HDIM / 128, BDIM), 256, 0, stream>>>(Pb, Vtb, out);
}
